// FocalDiceDsLoss_84464826843907
// MI455X (gfx1250) — matrix operand order repaired and checked by reading, not compiled, not run on hardware
//
#include <hip/hip_runtime.h>
#include <hip/hip_bf16.h>

typedef __attribute__((ext_vector_type(16))) _Float16 v16h;
typedef __attribute__((ext_vector_type(8)))  _Float16 v8h;
typedef __attribute__((ext_vector_type(8)))  float    v8f;

#define NCLS 33
#define KPAD 64
#define NROW 4096          // rows per batch
#define NBAT 2
#define ROWS (NBAT * NROW) // 8192
#define NTILE (NROW / 16)  // 256 row-tiles per batch

// -------- Phase 1: per-row softmax stats + f16 prob matrix ----------
// One thread per row. Deterministic block reduction (thread 0, fixed order).
__global__ void FDL_rows_kernel(const float* __restrict__ pred,
                                const int* __restrict__ targets,
                                const int* __restrict__ pchoice,
                                _Float16* __restrict__ P,
                                float* __restrict__ sq,
                                float* __restrict__ ce_part,
                                float* __restrict__ focal_part,
                                int* __restrict__ match_part) {
    __shared__ float sce[256];
    __shared__ float sfo[256];
    __shared__ int   smt[256];

    const int tid = threadIdx.x;
    const int row = blockIdx.x * 256 + tid;
    const float* x = pred + (size_t)row * NCLS;

    float xl[NCLS];
    float mx = -3.0e38f;
    #pragma unroll
    for (int c = 0; c < NCLS; ++c) { xl[c] = x[c]; mx = fmaxf(mx, xl[c]); }

    float el[NCLS];
    float s = 0.0f;
    #pragma unroll
    for (int c = 0; c < NCLS; ++c) { el[c] = __expf(xl[c] - mx); s += el[c]; }
    const float inv = 1.0f / s;

    _Float16* prow = P + (size_t)row * KPAD;
    float sqv = 0.0f;
    #pragma unroll
    for (int c = 0; c < NCLS; ++c) {
        _Float16 ph = (_Float16)(el[c] * inv);
        prow[c] = ph;
        float pf = (float)ph;          // use f16-rounded value so diag d2 ~ 0
        sqv += pf * pf;
    }
    #pragma unroll
    for (int c = NCLS; c < KPAD; ++c) prow[c] = (_Float16)0.0f;
    sq[row] = sqv;

    const int t = targets[row];
    const float logpt = xl[t] - mx - __logf(s);
    const float pn = __expf(logpt);
    const float om = 1.0f - pn;

    sce[tid] = -logpt;
    sfo[tid] = om * om;
    smt[tid] = (t == pchoice[row]) ? 1 : 0;
    __syncthreads();

    if (tid == 0) {
        float a = 0.0f, b = 0.0f; int m = 0;
        for (int i = 0; i < 256; ++i) { a += sce[i]; b += sfo[i]; m += smt[i]; }
        ce_part[blockIdx.x]    = a;
        focal_part[blockIdx.x] = b;
        match_part[blockIdx.x] = m;
    }
}

// -------- Phase 2: pairwise distance sum via WMMA P*P^T tiles ----------
// grid = (256 i-tiles, 2 batches), block = 128 (4 waves). Each wave owns row
// tile i0, strides over j-tiles. Two v_wmma_f32_16x16x32_f16 per 16x16 tile
// (K padded 33->64). Epilogue: sum sqrt(max(sq_i+sq_j-2*dot, 0)) using the
// raw hardware v_sqrt_f32 (1-instruction TRANS op, co-executes with VALU).
__global__ void FDL_pair_kernel(const _Float16* __restrict__ P,
                                const float* __restrict__ sq,
                                float* __restrict__ dist_part) {
    __shared__ float red[128];
    const int tid  = threadIdx.x;
    const int lane = tid & 31;
    const int wave = tid >> 5;
    const int i0   = blockIdx.x;   // 0..255
    const int b    = blockIdx.y;   // 0..1
    const int rbase = b * NROW;
    const int l15  = lane & 15;
    const int g    = lane >> 4;    // half-wave group

    union U { v16h v; v8h h[2]; };

    // A operand (16x32 f16, M = lane&15): halfs i=0..7 -> K=8g+i ; i=8..15 -> K=16+8g+(i-8)
    const _Float16* arow = P + (size_t)(rbase + i0 * 16 + l15) * KPAD;
    U a0, a1;
    a0.h[0] = *(const v8h*)(arow +  0 + 8 * g);
    a0.h[1] = *(const v8h*)(arow + 16 + 8 * g);
    a1.h[0] = *(const v8h*)(arow + 32 +  0 + 8 * g);
    a1.h[1] = *(const v8h*)(arow + 32 + 16 + 8 * g);

    // sq_i for acc element r: local row M = r + 8*g
    float sqi[8];
    #pragma unroll
    for (int r = 0; r < 8; ++r) sqi[r] = sq[rbase + i0 * 16 + r + 8 * g];

    float local = 0.0f;
    for (int j0 = wave; j0 < NTILE; j0 += 4) {
        // B operand (32x16 f16, N = lane&15): halfs i=0..7 -> K=8g+i ; i=8..15 -> K=16+8g+(i-8)
        const _Float16* brow = P + (size_t)(rbase + j0 * 16 + l15) * KPAD;
        U b0, b1;
        b0.h[0] = *(const v8h*)(brow +      8 * g);
        b0.h[1] = *(const v8h*)(brow + 16 + 8 * g);
        b1.h[0] = *(const v8h*)(brow + 32 +      8 * g);
        b1.h[1] = *(const v8h*)(brow + 32 + 16 + 8 * g);

        v8f acc = {};
        acc = __builtin_amdgcn_wmma_f32_16x16x32_f16(false, a0.v, false, b0.v,
                                                     (short)0, acc, false, false);
        acc = __builtin_amdgcn_wmma_f32_16x16x32_f16(false, a1.v, false, b1.v,
                                                     (short)0, acc, false, false);

        const float sqj = sq[rbase + j0 * 16 + l15];
        #pragma unroll
        for (int r = 0; r < 8; ++r) {
            float d2 = sqi[r] + sqj - 2.0f * acc[r];
            // raw hardware sqrt (v_sqrt_f32), branchless select
            float d = __builtin_amdgcn_sqrtf(d2);
            local += (d2 > 0.0f) ? d : 0.0f;
        }
    }

    red[tid] = local;
    __syncthreads();
    if (tid == 0) {
        float ssum = 0.0f;
        for (int i = 0; i < 128; ++i) ssum += red[i];
        dist_part[b * NTILE + i0] = ssum;
    }
}

// -------- Phase 3: combine scalars (fixed-order, single thread) ----------
__global__ void FDL_final_kernel(const float* __restrict__ ce_part,
                                 const float* __restrict__ focal_part,
                                 const int* __restrict__ match_part,
                                 const float* __restrict__ dist_part,
                                 float* __restrict__ out) {
    if (threadIdx.x == 0 && blockIdx.x == 0) {
        float ce = 0.0f, fo = 0.0f; int m = 0;
        for (int i = 0; i < 32; ++i) { ce += ce_part[i]; fo += focal_part[i]; m += match_part[i]; }
        float S = 0.0f;
        for (int i = 0; i < 2 * NTILE; ++i) S += dist_part[i];

        const float inv_bn = 1.0f / (float)ROWS;
        const float ce_mean = ce * inv_bn;
        const float focal_mean = fo * inv_bn;
        const float loss = ce_mean * focal_mean;   // mean((1-p)^2 * ce_scalar)

        const float total = (float)ROWS;
        const float dice = 1.0f - 2.0f * (((float)m + 1.0f) / (2.0f * total + 1.0f));

        // dis_map == 0.05 everywhere (one-hot distances are 0 or sqrt(2))
        const float meanD = S / ((float)NBAT * (float)NROW * (float)NROW);
        const float dml = 0.05f * logf(0.05f) - 0.05f * meanD;

        out[0] = loss + dice + dml;
    }
}

extern "C" void kernel_launch(void* const* d_in, const int* in_sizes, int n_in,
                              void* d_out, int out_size, void* d_ws, size_t ws_size,
                              hipStream_t stream) {
    const float* pred    = (const float*)d_in[0];
    // d_in[1] (y_ohe) is analytically unused: dis_map collapses to constant 0.05
    const int*   targets = (const int*)d_in[2];
    const int*   pchoice = (const int*)d_in[3];

    char* ws = (char*)d_ws;
    size_t off = 0;
    _Float16* P = (_Float16*)(ws + off);  off += (size_t)ROWS * KPAD * sizeof(_Float16); // 1 MiB
    float* sq         = (float*)(ws + off); off += (size_t)ROWS * sizeof(float);
    float* ce_part    = (float*)(ws + off); off += 32 * sizeof(float);
    float* focal_part = (float*)(ws + off); off += 32 * sizeof(float);
    int*   match_part = (int*)  (ws + off); off += 32 * sizeof(int);
    float* dist_part  = (float*)(ws + off); off += 2 * NTILE * sizeof(float);

    FDL_rows_kernel<<<ROWS / 256, 256, 0, stream>>>(pred, targets, pchoice, P, sq,
                                                    ce_part, focal_part, match_part);
    FDL_pair_kernel<<<dim3(NTILE, NBAT), 128, 0, stream>>>(P, sq, dist_part);
    FDL_final_kernel<<<1, 32, 0, stream>>>(ce_part, focal_part, match_part, dist_part,
                                           (float*)d_out);
}
